// ImprovedSelfAttention_12103217840547
// MI455X (gfx1250) — matrix operand order repaired and checked by reading, not compiled, not run on hardware
//
#include <hip/hip_runtime.h>

// ---------------------------------------------------------------------------
// ImprovedSelfAttention forward for MI455X (gfx1250, wave32, WMMA bf16,
// async global->LDS staging with ASYNCcnt double buffering)
// B=4, N=2048, D=320, H=10, DH=128, INNER=1280
// ---------------------------------------------------------------------------

#define SEQ    2048
#define DMODEL 320
#define NHEAD  10
#define DHEAD  128
#define INNER  1280   // NHEAD * DHEAD
#define ROWS   8192   // B * SEQ

typedef __attribute__((ext_vector_type(16))) __bf16 v16bf;
typedef __attribute__((ext_vector_type(8)))  float  v8f;

union ABFrag { v16bf v; unsigned int u[8]; };

__device__ __forceinline__ unsigned short f2bf(float f) {
  unsigned int u = __float_as_uint(f);
  u += 0x7fffu + ((u >> 16) & 1u);          // round-to-nearest-even
  return (unsigned short)(u >> 16);
}
__device__ __forceinline__ unsigned int pack2(float a, float b) {
  return (unsigned int)f2bf(a) | ((unsigned int)f2bf(b) << 16);
}
// A-operand (16x32 bf16) per-lane K map: pair p, lane-half h
__device__ __forceinline__ int a_kmap(int p, int h) {
  int e = 2 * p;
  return (e < 8) ? (e + 8 * h) : (e + 8 + 8 * h);
}
// B-operand (32x16 bf16) per-lane K map
__device__ __forceinline__ int b_kmap(int p, int h) { return a_kmap(p, h); }

// Async copy 4 bytes global -> LDS (per-lane addresses), tracked by ASYNCcnt.
// GVS addressing: uniform 64-bit SGPR base + per-lane 32-bit byte offset.
__device__ __forceinline__ void async_b32(unsigned lds_off, unsigned voff,
                                          const void* sbase) {
  asm volatile("global_load_async_to_lds_b32 %0, %1, %2"
               :: "v"(lds_off), "v"(voff),
                  "s"((unsigned long long)(uintptr_t)sbase)
               : "memory");
}
__device__ __forceinline__ void wait_async16() {
  asm volatile("s_wait_asynccnt 0x10" ::: "memory");
}
__device__ __forceinline__ void wait_async0() {
  asm volatile("s_wait_asynccnt 0x0" ::: "memory");
}

// ---------------------------------------------------------------------------
// Generic tiled GEMM:  C = A[M,K] * Bw[K,N]
// MODE 0: A = fp32 x; writes bf16 Q (scaled, [row][INNER]),
//         K ([b,h][n][dh]) and V-transposed ([b,h][dh][n]) workspaces.
// MODE 1: A = bf16 attn-out; writes fp32 d_out + bias.
// Block tile 128x64, BK=32, 8 waves, each wave 32x32 (2x2 WMMA tiles)
// ---------------------------------------------------------------------------
template <int MODE>
__global__ __launch_bounds__(256) void wmma_gemm(
    const void* __restrict__ Aptr, const float* __restrict__ Bw,
    void* __restrict__ Cq, unsigned short* __restrict__ Ck,
    unsigned short* __restrict__ Cvt, const float* __restrict__ bias,
    int N, int K)
{
  __shared__ unsigned short As[128 * 34];  // [m][k], padded stride
  __shared__ unsigned short Bs[64 * 34];   // [n][k] (transposed), padded

  const int tid  = threadIdx.x;
  const int wave = tid >> 5, lane = tid & 31;
  const int half = lane >> 4, l16 = lane & 15;
  const int wm = (wave & 3) * 32;
  const int wn = (wave >> 2) * 32;
  const int m0 = blockIdx.y * 128;
  const int n0 = blockIdx.x * 64;

  v8f acc[2][2] = {};

  for (int k0 = 0; k0 < K; k0 += 32) {
    __syncthreads();
    if (MODE == 0) {                       // fp32 A -> bf16 LDS
      const float* A = (const float*)Aptr;
      #pragma unroll
      for (int i = 0; i < 4; ++i) {
        int flat = i * 256 + tid;          // 1024 float4
        int row = flat >> 3, c4 = flat & 7;
        const float4 f = *(const float4*)(A + (size_t)(m0 + row) * K + k0 + c4 * 4);
        unsigned int* dst = (unsigned int*)&As[row * 34 + c4 * 4];
        dst[0] = pack2(f.x, f.y);
        dst[1] = pack2(f.z, f.w);
      }
    } else {                               // bf16 A -> LDS copy
      const unsigned short* A = (const unsigned short*)Aptr;
      #pragma unroll
      for (int i = 0; i < 8; ++i) {
        int flat = i * 256 + tid;          // 2048 uints
        int row = flat >> 4, cu = flat & 15;
        *(unsigned int*)&As[row * 34 + cu * 2] =
            *(const unsigned int*)(A + (size_t)(m0 + row) * K + k0 + cu * 2);
      }
    }
    // B block [k0..k0+32) x [n0..n0+64), stored transposed Bs[n][k]
    #pragma unroll
    for (int i = 0; i < 2; ++i) {
      int flat = i * 256 + tid;            // 512 float4
      int kr = flat >> 4, c4 = flat & 15;
      const float4 f = *(const float4*)(Bw + (size_t)(k0 + kr) * N + n0 + c4 * 4);
      Bs[(c4 * 4 + 0) * 34 + kr] = f2bf(f.x);
      Bs[(c4 * 4 + 1) * 34 + kr] = f2bf(f.y);
      Bs[(c4 * 4 + 2) * 34 + kr] = f2bf(f.z);
      Bs[(c4 * 4 + 3) * 34 + kr] = f2bf(f.w);
    }
    __syncthreads();

    ABFrag afr[2], bfr[2];
    #pragma unroll
    for (int mt = 0; mt < 2; ++mt)
      #pragma unroll
      for (int p = 0; p < 8; ++p)
        afr[mt].u[p] = *(const unsigned int*)&As[(wm + mt * 16 + l16) * 34 + a_kmap(p, half)];
    #pragma unroll
    for (int nt = 0; nt < 2; ++nt)
      #pragma unroll
      for (int p = 0; p < 8; ++p)
        bfr[nt].u[p] = *(const unsigned int*)&Bs[(wn + nt * 16 + l16) * 34 + b_kmap(p, half)];

    #pragma unroll
    for (int mt = 0; mt < 2; ++mt)
      #pragma unroll
      for (int nt = 0; nt < 2; ++nt)
        acc[mt][nt] = __builtin_amdgcn_wmma_f32_16x16x32_bf16(
            false, afr[mt].v, false, bfr[nt].v, (short)0, acc[mt][nt], false, false);
  }

  #pragma unroll
  for (int mt = 0; mt < 2; ++mt)
    #pragma unroll
    for (int nt = 0; nt < 2; ++nt)
      #pragma unroll
      for (int r = 0; r < 8; ++r) {
        const int row = m0 + wm + mt * 16 + r + 8 * half;   // C layout: M = r + 8*half
        const int col = n0 + wn + nt * 16 + l16;
        float v = acc[mt][nt][r];
        if (MODE == 0) {
          const int b = row >> 11, n = row & (SEQ - 1);
          if (col < INNER) {                                // Q, pre-scaled by DH^-0.5
            ((unsigned short*)Cq)[(size_t)row * INNER + col] =
                f2bf(v * 0.08838834764831845f);
          } else if (col < 2 * INNER) {                     // K -> [b,h][n][dh]
            const int cc = col - INNER, hh = cc >> 7, dh = cc & 127;
            Ck[((size_t)(b * NHEAD + hh) << 18) + (n << 7) + dh] = f2bf(v);
          } else {                                          // V -> transposed [b,h][dh][n]
            const int cc = col - 2 * INNER, hh = cc >> 7, dh = cc & 127;
            Cvt[((size_t)(b * NHEAD + hh) << 18) + (dh << 11) + n] = f2bf(v);
          }
        } else {
          ((float*)Cq)[(size_t)row * N + col] = v + bias[col];
        }
      }
}

// ---------------------------------------------------------------------------
// Flash attention: one block per (b, h, 128-query tile). 8 waves x 16 q-rows.
// K/V tiles staged into double-buffered LDS with async global->LDS copies.
// ---------------------------------------------------------------------------
__global__ __launch_bounds__(256) void flash_attn(
    const unsigned short* __restrict__ q_ws,   // bf16 [ROWS][INNER], pre-scaled
    const unsigned short* __restrict__ k_ws,   // bf16 [B*H][SEQ][DHEAD]
    const unsigned short* __restrict__ vt_ws,  // bf16 [B*H][DHEAD][SEQ]
    unsigned short* __restrict__ attnout)      // bf16 [ROWS][INNER]
{
  __shared__ unsigned short Ks[2][32 * 130];   // [key][dh]  == B-frag layout (QK^T)
  __shared__ unsigned short Vt[2][128 * 34];   // [dh][key]  == B-frag layout (PV)
  __shared__ unsigned short Pb[8 * 16 * 34];   // per-wave P tile [16][32]

  const int tid = threadIdx.x, wave = tid >> 5, lane = tid & 31;
  const int half = lane >> 4, l16 = lane & 15;
  const int idx  = blockIdx.x;
  const int qblk = idx & 15;
  const int h    = (idx >> 4) % NHEAD;
  const int b    = idx / (16 * NHEAD);

  const size_t brow = (size_t)b * SEQ;
  const int qbase = qblk * 128 + wave * 16;
  const unsigned short* kbase = k_ws + ((size_t)(b * NHEAD + h) << 18);
  const unsigned short* vbase = vt_ws + ((size_t)(b * NHEAD + h) << 18);

  // Per-lane constant byte offsets for the async staging (only the uniform
  // SGPR base advances with the key-block index).
  unsigned kvoff[8], kldo[8], vvoff[8], vldo[8];
  #pragma unroll
  for (int i = 0; i < 8; ++i) {
    const int flat = i * 256 + tid;            // 2048 B32 copies per tile
    const int key = flat >> 6, cu = flat & 63; // K: [key][dh pair]
    kvoff[i] = (unsigned)((key * DHEAD + cu * 2) * 2);
    kldo[i]  = (unsigned)(uintptr_t)&Ks[0][key * 130 + cu * 2];
    const int dh = flat >> 4, cv = flat & 15;  // Vt: [dh][key pair]
    vvoff[i] = (unsigned)((dh * SEQ + cv * 2) * 2);
    vldo[i]  = (unsigned)(uintptr_t)&Vt[0][dh * 34 + cv * 2];
  }
  const unsigned ksz = (unsigned)sizeof(Ks[0]);
  const unsigned vsz = (unsigned)sizeof(Vt[0]);

  auto issue_tile = [&](int buf, int kb) {
    const unsigned short* kb_base = kbase + (size_t)kb * 32 * DHEAD;
    const unsigned short* vb_base = vbase + (size_t)kb * 32;
    const unsigned kadd = buf ? ksz : 0u;
    const unsigned vadd = buf ? vsz : 0u;
    #pragma unroll
    for (int i = 0; i < 8; ++i) async_b32(kldo[i] + kadd, kvoff[i], kb_base);
    #pragma unroll
    for (int i = 0; i < 8; ++i) async_b32(vldo[i] + vadd, vvoff[i], vb_base);
  };

  // Q fragments for this wave's 16 rows (K = DHEAD = 4 x 32)
  ABFrag qf[4];
  #pragma unroll
  for (int kk = 0; kk < 4; ++kk)
    #pragma unroll
    for (int p = 0; p < 8; ++p)
      qf[kk].u[p] = *(const unsigned int*)(
          q_ws + (brow + qbase + l16) * INNER + h * DHEAD + kk * 32 + a_kmap(p, half));

  v8f O[8] = {};
  float mrow[8], lrow[8];
  #pragma unroll
  for (int r = 0; r < 8; ++r) { mrow[r] = -1e30f; lrow[r] = 0.f; }

  constexpr int NB = SEQ / 32;
  issue_tile(0, 0);

  for (int kb = 0; kb < NB; ++kb) {
    const int cur = kb & 1;
    if (kb + 1 < NB) {                       // prefetch next tile, then drain cur
      issue_tile(cur ^ 1, kb + 1);
      wait_async16();                        // in-order: first 16 (cur tile) done
    } else {
      wait_async0();
    }
    __syncthreads();                         // all waves' portions landed

    // S(16x32) = Q(16x128) * Kblk^T
    v8f s[2] = {};
    #pragma unroll
    for (int kk = 0; kk < 4; ++kk)
      #pragma unroll
      for (int nt = 0; nt < 2; ++nt) {
        ABFrag bf;
        #pragma unroll
        for (int p = 0; p < 8; ++p)
          bf.u[p] = *(const unsigned int*)&Ks[cur][(nt * 16 + l16) * 130 + kk * 32 + b_kmap(p, half)];
        s[nt] = __builtin_amdgcn_wmma_f32_16x16x32_bf16(
            false, qf[kk].v, false, bf.v, (short)0, s[nt], false, false);
      }

    // online softmax (row m = r + 8*half lives within one 16-lane half)
    float scale[8];
    #pragma unroll
    for (int r = 0; r < 8; ++r) {
      float mx = fmaxf(s[0][r], s[1][r]);
      #pragma unroll
      for (int msk = 8; msk >= 1; msk >>= 1) mx = fmaxf(mx, __shfl_xor(mx, msk, 32));
      const float mnew = fmaxf(mrow[r], mx);
      scale[r] = __expf(mrow[r] - mnew);
      mrow[r] = mnew;
      float p0 = __expf(s[0][r] - mnew);
      float p1 = __expf(s[1][r] - mnew);
      s[0][r] = p0; s[1][r] = p1;
      float rs = p0 + p1;
      #pragma unroll
      for (int msk = 8; msk >= 1; msk >>= 1) rs += __shfl_xor(rs, msk, 32);
      lrow[r] = lrow[r] * scale[r] + rs;
    }
    #pragma unroll
    for (int t = 0; t < 8; ++t)
      #pragma unroll
      for (int r = 0; r < 8; ++r) O[t][r] *= scale[r];

    // C-layout P -> row-major LDS (wave-private, DS in-order) -> A-fragment
    unsigned short* P = &Pb[wave * 16 * 34];
    #pragma unroll
    for (int r = 0; r < 8; ++r) {
      const int row = r + 8 * half;
      P[row * 34 + l16]      = f2bf(s[0][r]);
      P[row * 34 + 16 + l16] = f2bf(s[1][r]);
    }
    ABFrag pf;
    #pragma unroll
    for (int p = 0; p < 8; ++p)
      pf.u[p] = *(const unsigned int*)&P[l16 * 34 + a_kmap(p, half)];

    // O(16x128) += P(16x32) * Vblk(32x128)
    #pragma unroll
    for (int nt = 0; nt < 8; ++nt) {
      ABFrag bf;
      #pragma unroll
      for (int p = 0; p < 8; ++p)
        bf.u[p] = *(const unsigned int*)&Vt[cur][(nt * 16 + l16) * 34 + b_kmap(p, half)];
      O[nt] = __builtin_amdgcn_wmma_f32_16x16x32_bf16(
          false, pf.v, false, bf.v, (short)0, O[nt], false, false);
    }
    __syncthreads();                         // tile consumed; buffer reusable
  }

  // finalize: O /= l, store bf16 to [ROWS][INNER]
  #pragma unroll
  for (int r = 0; r < 8; ++r) {
    const float inv = 1.0f / lrow[r];
    #pragma unroll
    for (int nt = 0; nt < 8; ++nt) {
      const int row = qbase + r + 8 * half;
      const int col = h * DHEAD + nt * 16 + l16;
      attnout[(brow + row) * INNER + col] = f2bf(O[nt][r] * inv);
    }
  }
}

// ---------------------------------------------------------------------------
extern "C" void kernel_launch(void* const* d_in, const int* in_sizes, int n_in,
                              void* d_out, int out_size, void* d_ws, size_t ws_size,
                              hipStream_t stream) {
  (void)in_sizes; (void)n_in; (void)out_size; (void)ws_size;
  const float* x     = (const float*)d_in[0];   // [4,2048,320]
  const float* w_qkv = (const float*)d_in[1];   // [320,3840]
  const float* w_out = (const float*)d_in[2];   // [1280,320]
  const float* b_out = (const float*)d_in[3];   // [320]
  float* out = (float*)d_out;                   // [4,2048,320]

  const size_t PLANE = (size_t)ROWS * INNER;    // 10.49M bf16 elements
  unsigned short* q_ws    = (unsigned short*)d_ws;        // [ROWS][INNER]
  unsigned short* k_ws    = q_ws + PLANE;                 // [B*H][SEQ][DHEAD]
  unsigned short* vt_ws   = k_ws + PLANE;                 // [B*H][DHEAD][SEQ]
  unsigned short* attn_ws = vt_ws + PLANE;                // [ROWS][INNER]  (~84MB total)

  dim3 blk(256);
  // 1) qkv = x @ w_qkv; Q scaled; K head-major; V transposed
  wmma_gemm<0><<<dim3(3 * INNER / 64, ROWS / 128), blk, 0, stream>>>(
      x, w_qkv, q_ws, k_ws, vt_ws, nullptr, 3 * INNER, DMODEL);
  // 2) flash attention per (b, h, 128-query tile)
  flash_attn<<<dim3(4 * NHEAD * (SEQ / 128)), blk, 0, stream>>>(
      q_ws, k_ws, vt_ws, attn_ws);
  // 3) out = attn @ w_out + b_out, fp32 out
  wmma_gemm<1><<<dim3(DMODEL / 64, ROWS / 128), blk, 0, stream>>>(
      attn_ws, w_out, out, nullptr, nullptr, b_out, DMODEL, INNER);
}
